// TextGCN_3985729651214
// MI455X (gfx1250) — matrix operand order repaired and checked by reading, not compiled, not run on hardware
//
#include <hip/hip_runtime.h>
#include <hip/hip_bf16.h>
#include <stdint.h>

// ---------------------------------------------------------------------------
// Types for CDNA5 WMMA
// ---------------------------------------------------------------------------
typedef __bf16 bf16_t;
typedef __attribute__((ext_vector_type(16))) __bf16 v16bf;
typedef __attribute__((ext_vector_type(8)))  float  v8f;

union Frag16 { v16bf v; uint32_t u[8]; };

#define TM 128
#define TN 128
#define TK 64
#define KPAD 8   // 72 bf16 per LDS row = 144 B, multiple of 16 B (keeps b128 ops aligned)

#define AS1 __attribute__((address_space(1)))
#define AS3 __attribute__((address_space(3)))

// 16-byte chunk type matching the async-copy builtin's parameter type
// (diagnostic showed: "__attribute__((__vector_size__(4 * sizeof(int)))) int __device__ *")
typedef int v4i_ld __attribute__((vector_size(16)));

// gfx1250 async global->LDS copies (ASYNCcnt-tracked), guarded per toolchain.
#if defined(__AMDGCN__) && __has_builtin(__builtin_amdgcn_global_load_async_to_lds_b128)
#define HAVE_ASYNC_LDS 1
#else
#define HAVE_ASYNC_LDS 0
#endif

__device__ __forceinline__ void wait_async_copies() {
#if HAVE_ASYNC_LDS
#if __has_builtin(__builtin_amdgcn_s_wait_asynccnt)
  __builtin_amdgcn_s_wait_asynccnt(0);
#else
  asm volatile("s_wait_asynccnt 0x0" ::: "memory");
#endif
#endif
}

__device__ __forceinline__ bf16_t f2bf(float f) {
  union { float f; uint32_t u; } x; x.f = f;
  uint32_t r = x.u + 0x7FFFu + ((x.u >> 16) & 1u);   // round-to-nearest-even
  uint16_t h = (uint16_t)(r >> 16);
  bf16_t o; __builtin_memcpy(&o, &h, 2);
  return o;
}

// ---------------------------------------------------------------------------
// f32 -> bf16 cast (row-major preserved)
// ---------------------------------------------------------------------------
__global__ __launch_bounds__(256) void cast_f32_bf16(const float* __restrict__ in,
                                                     bf16_t* __restrict__ out, int n) {
  int i = (blockIdx.x * 256 + threadIdx.x) * 4;
  if (i + 3 < n) {
    float4 v = *(const float4*)(in + i);
    out[i + 0] = f2bf(v.x);
    out[i + 1] = f2bf(v.y);
    out[i + 2] = f2bf(v.z);
    out[i + 3] = f2bf(v.w);
  }
}

// ---------------------------------------------------------------------------
// f32 -> bf16 cast + transpose: out[c][r] = in[r][c].  in: [R][C], out: [C][R]
// 32x32 LDS tile, coalesced on both sides.
// ---------------------------------------------------------------------------
__global__ __launch_bounds__(256) void cast_transpose_f32_bf16(
    const float* __restrict__ in, bf16_t* __restrict__ out, int R, int C) {
  __shared__ bf16_t tile[32][33];
  const int bx = blockIdx.x * 32;               // column base (of in)
  const int by = blockIdx.y * 32;               // row base (of in)
  const int tx = threadIdx.x & 31;
  const int ty = threadIdx.x >> 5;              // 0..7
  #pragma unroll
  for (int i = 0; i < 32; i += 8)
    tile[ty + i][tx] = f2bf(in[(size_t)(by + ty + i) * C + (bx + tx)]);
  __syncthreads();
  #pragma unroll
  for (int i = 0; i < 32; i += 8)
    out[(size_t)(bx + ty + i) * R + (by + tx)] = tile[tx][ty + i];
}

// ---------------------------------------------------------------------------
// Stage one 128x64 bf16 tile into LDS (16B chunks, contiguous along K).
// Async path overlaps with WMMA compute; fallback is plain load/store.
// ---------------------------------------------------------------------------
__device__ __forceinline__ void stage_tile(const bf16_t* __restrict__ g,
                                           bf16_t (*lds)[TK + KPAD],
                                           int base, int K, int k0, int tid) {
  #pragma unroll
  for (int i = 0; i < (TM * TK / 8) / 256; i++) {
    int gi = tid + i * 256;                      // 0..1023
    int r  = gi >> 3;                            // row within tile
    int ko = (gi & 7) << 3;                      // k offset (multiple of 8)
    const bf16_t* src = g + (size_t)(base + r) * K + k0 + ko;
#if HAVE_ASYNC_LDS
    __builtin_amdgcn_global_load_async_to_lds_b128(
        (AS1 v4i_ld*)src, (AS3 v4i_ld*)&lds[r][ko], 0, 0);
#else
    uint4 d = *(const uint4*)src;
    *(uint4*)&lds[r][ko] = d;
#endif
  }
}

// ---------------------------------------------------------------------------
// Tiled bf16 WMMA GEMM: C[M,N] = A @ Bt^T, f32 accumulate.
//   A  : [M][K] row-major bf16 (K contiguous)
//   Bt : [N][K] row-major bf16 (K contiguous)  == B[K][N] transposed
// Epilogue modes:
//   0: C = acc
//   1: C = relu(acc)
//   2: C = relu(acc) * mask[row]
//   3: C = res[row,col] + relu(acc) * mask[row]
//   4: C = relu(acc + bias[col])
// Double-buffered LDS; async copies for tile i+1 issued before WMMA on tile i.
// ---------------------------------------------------------------------------
__global__ __launch_bounds__(256) void gemm_wmma_bf16(
    const bf16_t* __restrict__ A, const bf16_t* __restrict__ Bt,
    float* __restrict__ C,
    const float* __restrict__ res, const float* __restrict__ bias,
    const float* __restrict__ mask,
    int M, int N, int K, int mode)
{
  __shared__ bf16_t As[2][TM][TK + KPAD];
  __shared__ bf16_t Bs[2][TN][TK + KPAD];

  const int tid   = threadIdx.x;
  const int lane  = tid & 31;
  const int wave  = tid >> 5;            // 8 waves
  const int l16   = lane & 15;
  const int hi    = lane >> 4;           // half-wave select (0/1)
  const int waveM = wave & 1;            // 2 waves along M (64 rows each)
  const int waveN = wave >> 1;           // 4 waves along N (32 cols each)

  const int m0 = blockIdx.y * TM;
  const int n0 = blockIdx.x * TN;

  v8f acc[4][2];
  #pragma unroll
  for (int i = 0; i < 4; i++)
    #pragma unroll
    for (int j = 0; j < 2; j++)
      #pragma unroll
      for (int r = 0; r < 8; r++) acc[i][j][r] = 0.f;

  int buf = 0;
  stage_tile(A,  As[0], m0, K, 0, tid);
  stage_tile(Bt, Bs[0], n0, K, 0, tid);
  wait_async_copies();
  __syncthreads();

  for (int k0 = 0; k0 < K; k0 += TK) {
    const int nb = buf ^ 1;
    if (k0 + TK < K) {                       // prefetch next tile while computing
      stage_tile(A,  As[nb], m0, K, k0 + TK, tid);
      stage_tile(Bt, Bs[nb], n0, K, k0 + TK, tid);
    }

    #pragma unroll
    for (int kk = 0; kk < TK; kk += 32) {
      Frag16 af[4], bfg[2];
      #pragma unroll
      for (int mt = 0; mt < 4; mt++) {
        int m = waveM * 64 + mt * 16 + l16;
        const bf16_t* srow = &As[buf][m][kk + (hi ? 8 : 0)];
        #pragma unroll
        for (int r = 0; r < 8; r++) {
          int k = ((r & 4) ? 16 : 0) + 2 * (r & 3);     // A 16-bit WMMA layout
          af[mt].u[r] = *(const uint32_t*)(srow + k);
        }
      }
      #pragma unroll
      for (int nt = 0; nt < 2; nt++) {
        int n = waveN * 32 + nt * 16 + l16;
        const bf16_t* srow = &Bs[buf][n][kk + (hi ? 8 : 0)];
        #pragma unroll
        for (int r = 0; r < 8; r++)
          bfg[nt].u[r] = *(const uint32_t*)(srow + ((r & 4) ? 16 : 0) + 2 * (r & 3)); // B 16-bit WMMA layout
      }
      #pragma unroll
      for (int mt = 0; mt < 4; mt++)
        #pragma unroll
        for (int nt = 0; nt < 2; nt++)
          acc[mt][nt] = __builtin_amdgcn_wmma_f32_16x16x32_bf16(
              false, af[mt].v, false, bfg[nt].v,
              (short)0, acc[mt][nt], false, false);
    }

    wait_async_copies();
    __syncthreads();
    buf = nb;
  }

  // ---- fused epilogue ----
  #pragma unroll
  for (int mt = 0; mt < 4; mt++) {
    #pragma unroll
    for (int nt = 0; nt < 2; nt++) {
      int col = n0 + waveN * 32 + nt * 16 + l16;
      #pragma unroll
      for (int r = 0; r < 8; r++) {
        int row = m0 + waveM * 64 + mt * 16 + r + hi * 8;  // C layout: VGPR r -> M=r(+8)
        float v = acc[mt][nt][r];
        size_t idx = (size_t)row * N + col;
        if (mode == 1)      v = fmaxf(v, 0.f);
        else if (mode == 2) v = fmaxf(v, 0.f) * mask[row];
        else if (mode == 3) v = res[idx] + fmaxf(v, 0.f) * mask[row];
        else if (mode == 4) v = fmaxf(v + bias[col], 0.f);
        C[idx] = v;
      }
    }
  }
}

// ---------------------------------------------------------------------------
// soft_att[row] = sigmoid(dot(X[row,:], att_w) + att_b)
// ---------------------------------------------------------------------------
__global__ __launch_bounds__(256) void att_sigmoid(const float* __restrict__ X,
                                                   const float* __restrict__ w,
                                                   const float* __restrict__ b,
                                                   float* __restrict__ satt, int D) {
  int row = blockIdx.x;
  float s = 0.f;
  for (int i = threadIdx.x; i < D; i += 256) s += X[(size_t)row * D + i] * w[i];
  __shared__ float red[256];
  red[threadIdx.x] = s; __syncthreads();
  for (int st = 128; st > 0; st >>= 1) {
    if (threadIdx.x < st) red[threadIdx.x] += red[threadIdx.x + st];
    __syncthreads();
  }
  if (threadIdx.x == 0) satt[row] = 1.f / (1.f + __expf(-(red[0] + b[0])));
}

// ---------------------------------------------------------------------------
// g = satt * H * mask;  ge[n,d] = sum_s(g) * max_s(g)
// ---------------------------------------------------------------------------
__global__ __launch_bounds__(256) void gate_pool(const float* __restrict__ H,
                                                 const float* __restrict__ satt,
                                                 const float* __restrict__ mask,
                                                 float* __restrict__ ge,
                                                 int seq, int D) {
  int d = blockIdx.x * 256 + threadIdx.x;
  int n = blockIdx.y;
  float s = 0.f, mx = -3.402823466e38f;
  for (int t = 0; t < seq; t++) {
    int row = n * seq + t;
    float g = satt[row] * H[(size_t)row * D + d] * mask[row];
    s += g;
    mx = fmaxf(mx, g);
  }
  ge[(size_t)n * D + d] = s * mx;
}

// ---------------------------------------------------------------------------
// out[n,c] = ge[n,:] . pred_w[:,c] + pred_b[c]
// ---------------------------------------------------------------------------
__global__ __launch_bounds__(256) void pred_head(const float* __restrict__ ge,
                                                 const float* __restrict__ pw,
                                                 const float* __restrict__ pb,
                                                 float* __restrict__ out,
                                                 int D, int C) {
  int n = blockIdx.x / C, c = blockIdx.x % C;
  float s = 0.f;
  for (int d = threadIdx.x; d < D; d += 256)
    s += ge[(size_t)n * D + d] * pw[(size_t)d * C + c];
  __shared__ float red[256];
  red[threadIdx.x] = s; __syncthreads();
  for (int st = 128; st > 0; st >>= 1) {
    if (threadIdx.x < st) red[threadIdx.x] += red[threadIdx.x + st];
    __syncthreads();
  }
  if (threadIdx.x == 0) out[blockIdx.x] = red[0] + pb[c];
}

// ---------------------------------------------------------------------------
extern "C" void kernel_launch(void* const* d_in, const int* in_sizes, int n_in,
                              void* d_out, int out_size, void* d_ws, size_t ws_size,
                              hipStream_t stream) {
  (void)in_sizes; (void)n_in; (void)out_size; (void)ws_size;

  const float* inX  = (const float*)d_in[0];   // [8,512,4096]
  const float* adj  = (const float*)d_in[1];   // [4096,4096]
  const float* mask = (const float*)d_in[2];   // [8,512,1] -> 4096 row mask
  const float* W0   = (const float*)d_in[3];
  const float* W1   = (const float*)d_in[4];
  const float* attw = (const float*)d_in[5];
  const float* attb = (const float*)d_in[6];
  const float* lnw  = (const float*)d_in[7];
  const float* lnb  = (const float*)d_in[8];
  const float* pw   = (const float*)d_in[9];
  const float* pb   = (const float*)d_in[10];
  float* out = (float*)d_out;

  const int Dm = 4096, NB = 8, SEQm = 512, NC = 20;
  const size_t E = (size_t)Dm * Dm;

  char* ws = (char*)d_ws;
  size_t off = 0;
  auto take = [&](size_t bytes) -> char* {
    char* p = ws + off;
    off += (bytes + 255) & ~(size_t)255;
    return p;
  };
  bf16_t* adjb = (bf16_t*)take(E * 2);
  bf16_t* opA  = (bf16_t*)take(E * 2);
  bf16_t* opB  = (bf16_t*)take(E * 2);
  float*  t1   = (float*)take(E * 4);
  float*  t2   = (float*)take(E * 4);
  float*  t3   = (float*)take(E * 4);
  float*  satt = (float*)take((size_t)Dm * 4);
  float*  ge   = (float*)take((size_t)NB * Dm * 4);

  dim3 gB(Dm / TN, Dm / TM);             // 32 x 32 macro-tiles
  dim3 castG((unsigned)(E / (256 * 4)));
  dim3 trG(Dm / 32, Dm / 32);

  // --- layer 0: X1 = relu(adj @ (W0^T @ X)) * mask ---
  cast_transpose_f32_bf16<<<trG, 256, 0, stream>>>(W0,  opA, Dm, Dm); // A  = W0^T
  cast_transpose_f32_bf16<<<trG, 256, 0, stream>>>(inX, opB, Dm, Dm); // Bt = X^T
  gemm_wmma_bf16<<<gB, 256, 0, stream>>>(opA, opB, t1, nullptr, nullptr, nullptr,
                                         Dm, Dm, Dm, 0);              // T0 -> t1
  cast_f32_bf16<<<castG, 256, 0, stream>>>(adj, adjb, (int)E);        // A = adj
  cast_transpose_f32_bf16<<<trG, 256, 0, stream>>>(t1, opB, Dm, Dm);  // Bt = T0^T
  gemm_wmma_bf16<<<gB, 256, 0, stream>>>(adjb, opB, t2, nullptr, nullptr, mask,
                                         Dm, Dm, Dm, 2);              // X1 -> t2

  // --- layer 1: X2 = X1 + relu(adj @ (W1^T @ X1)) * mask ---
  cast_transpose_f32_bf16<<<trG, 256, 0, stream>>>(W1, opA, Dm, Dm);  // A  = W1^T
  cast_transpose_f32_bf16<<<trG, 256, 0, stream>>>(t2, opB, Dm, Dm);  // Bt = X1^T
  gemm_wmma_bf16<<<gB, 256, 0, stream>>>(opA, opB, t1, nullptr, nullptr, nullptr,
                                         Dm, Dm, Dm, 0);              // T1 -> t1
  cast_transpose_f32_bf16<<<trG, 256, 0, stream>>>(t1, opB, Dm, Dm);  // Bt = T1^T
  gemm_wmma_bf16<<<gB, 256, 0, stream>>>(adjb, opB, t3, t2, nullptr, mask,
                                         Dm, Dm, Dm, 3);              // X2 -> t3

  // --- attention gate + linear head features ---
  att_sigmoid<<<Dm, 256, 0, stream>>>(t3, attw, attb, satt, Dm);
  cast_f32_bf16<<<castG, 256, 0, stream>>>(t3, opA, (int)E);          // A  = X2
  cast_transpose_f32_bf16<<<trG, 256, 0, stream>>>(lnw, opB, Dm, Dm); // Bt = lnw^T
  gemm_wmma_bf16<<<gB, 256, 0, stream>>>(opA, opB, t1, nullptr, lnb, nullptr,
                                         Dm, Dm, Dm, 4);              // H -> t1

  // --- gated sum*max pooling, prediction head ---
  dim3 gp(Dm / 256, NB);
  gate_pool<<<gp, 256, 0, stream>>>(t1, satt, mask, ge, SEQm, Dm);
  pred_head<<<NB * NC, 256, 0, stream>>>(ge, pw, pb, out, Dm, NC);
}
